// GraphCare_87316685128220
// MI455X (gfx1250) — matrix operand order repaired and checked by reading, not compiled, not run on hardware
//
#include <hip/hip_runtime.h>
#include <hip/hip_bf16.h>

typedef __attribute__((ext_vector_type(8)))  float  v8f;
typedef __attribute__((ext_vector_type(2)))  float  v2f;
typedef __attribute__((ext_vector_type(16))) __bf16 v16bf;

#define VOCAB 4096
#define HID   128
#define MV    20
#define NBAT  64
#define KSTEP 32

// ---- gfx1250 async copy global->LDS (ASYNCcnt path) ----
__device__ __forceinline__ void async_ld_b128(unsigned lds_off, const void* gaddr) {
  asm volatile("global_load_async_to_lds_b128 %0, %1, off"
               :: "v"(lds_off), "v"(gaddr) : "memory");
}
__device__ __forceinline__ void wait_async0() {
  asm volatile("s_wait_asynccnt 0" ::: "memory");
}

// ------------------------------------------------------------------
// Generic f32 WMMA NT-GEMM:  C = act( (A [+ (1+*epsp)*A2]) @ B^T + bias )
// A:[M,lda] row-major, B:[N,ldb] row-major (multiply by B^T).
// One 16x16 tile per wave, K stepped by 4 via v_wmma_f32_16x16x4_f32.
// HAS_A2 / RELU are compile-time so the hot loop has no EXEC juggling.
// ------------------------------------------------------------------
template <bool HAS_A2, bool RELU>
__global__ void gemm_f32_nt(const float* __restrict__ A, int lda,
                            const float* __restrict__ A2, const float* __restrict__ epsp,
                            const float* __restrict__ B, int ldb,
                            const float* __restrict__ bias,
                            float* __restrict__ C, int ldc,
                            int M, int N, int K)
{
  const int wave = threadIdx.x >> 5;
  const int lane = threadIdx.x & 31;
  const int half = lane >> 4;
  const int mtiles = (M + 15) >> 4;
  const int ntiles = (N + 15) >> 4;
  long tile = (long)blockIdx.x * (blockDim.x >> 5) + wave;
  if (tile >= (long)mtiles * ntiles) return;            // wave-uniform exit
  const int mt = (int)(tile / ntiles);
  const int nt = (int)(tile % ntiles);
  int mrow = mt * 16 + (lane & 15); if (mrow >= M) mrow = M - 1;  // clamp (stores guarded)
  int nrow = nt * 16 + (lane & 15); if (nrow >= N) nrow = N - 1;
  const float* arow  = A + (long)mrow * lda;
  const float* a2row = HAS_A2 ? (A2 + (long)mrow * lda) : nullptr;
  const float* brow  = B + (long)nrow * ldb;
  const float es = HAS_A2 ? (1.0f + epsp[0]) : 0.0f;
  v8f acc = {};
  for (int k = 0; k < K; k += 4) {
    const int ka = k + 2 * half;          // f32 A 16x4: lanes 0-15 K={0,1}, 16-31 K={2,3}
    v2f a, b;
    a.x = arow[ka];     a.y = arow[ka + 1];
    if constexpr (HAS_A2) {
      a.x = fmaf(es, a2row[ka], a.x);
      a.y = fmaf(es, a2row[ka + 1], a.y);
    }
    b.x = brow[ka];     b.y = brow[ka + 1];
    acc = __builtin_amdgcn_wmma_f32_16x16x4_f32(false, a, false, b, (short)0, acc,
                                                false, false);
  }
  const int n = nt * 16 + (lane & 15);
  if (n < N) {
    const float bn = bias ? bias[n] : 0.0f;
#pragma unroll
    for (int v = 0; v < 8; ++v) {
      const int m = mt * 16 + v + 8 * half;   // C/D: VGPR v holds rows v / v+8 per half
      if (m < M) {
        float val = acc[v] + bn;
        if (RELU) val = fmaxf(val, 0.0f);
        C[(long)m * ldc + n] = val;
      }
    }
  }
}

// ------------------------------------------------------------------
// Big attention GEMM, bf16 WMMA, f32 accumulate.
// C[1280,4096] = A[1280,4096] @ W[4096,4096]^T + bias
// grid = (32, 80), block = 256 (8 waves; all share the 16 A-rows of mt).
// A tile (16 x KSTEP fp32) staged per-block via async global->LDS DMA,
// double-buffered: DMA of step s+1 overlaps WMMA of step s.
// ------------------------------------------------------------------
__global__ void alpha_gemm_bf16(const float* __restrict__ A,
                                const float* __restrict__ W,
                                const float* __restrict__ bias,
                                float* __restrict__ C)
{
  __shared__ float sA[2][16 * KSTEP];     // 2 x 2KB double buffer
  const int t = threadIdx.x;
  const int wave = t >> 5;
  const int lane = t & 31;
  const int half = lane >> 4;
  const int nt = blockIdx.x * 8 + wave;   // 0..255
  const int mt = blockIdx.y;              // 0..79
  const float* Abase = A + (long)mt * 16 * VOCAB;                  // block-shared rows
  const float* brow  = W + (long)(nt * 16 + (lane & 15)) * VOCAB;  // wave-private rows

  // threads 0..127 each DMA one 16B chunk: row = t>>3, chunk = t&7
  const int srow = t >> 3, sch = t & 7;

  if (t < 128)
    async_ld_b128((unsigned)(uintptr_t)&sA[0][srow * KSTEP + sch * 4],
                  Abase + (long)srow * VOCAB + sch * 4);

  v8f acc = {};
  const int nsteps = VOCAB / KSTEP;       // 128
  for (int s = 0; s < nsteps; ++s) {
    const int cur = s & 1;
    wait_async0();                        // my DMAs (if any) have landed in LDS
    __syncthreads();                      // everyone's have landed
    if (s + 1 < nsteps && t < 128)        // overlap next tile's DMA with compute
      async_ld_b128((unsigned)(uintptr_t)&sA[cur ^ 1][srow * KSTEP + sch * 4],
                    Abase + (long)srow * VOCAB + (s + 1) * KSTEP + sch * 4);

    v16bf a, b;
    const float* sar = &sA[cur][(lane & 15) * KSTEP];
    const int ka = 8 * half;              // bf16 A 16x32: runs [8h,8h+8) and [16+8h,...)
#pragma unroll
    for (int i = 0; i < 8; ++i) {
      a[i]     = (__bf16)sar[ka + i];
      a[i + 8] = (__bf16)sar[ka + 16 + i];
    }
    const float* bp = brow + s * KSTEP + 8 * half;   // K runs [8h,8h+8) and [16+8h,...) of W row n
#pragma unroll
    for (int i = 0; i < 8; ++i) {
      b[i]     = (__bf16)bp[i];
      b[i + 8] = (__bf16)bp[16 + i];
    }
    __builtin_prefetch(bp + KSTEP, 0, 1);            // global_prefetch ahead in W row
    acc = __builtin_amdgcn_wmma_f32_16x16x32_bf16(false, a, false, b, (short)0, acc,
                                                  false, false);
    __syncthreads();                      // all waves done reading sA[cur]
  }
  const int n = nt * 16 + (lane & 15);
  const float bn = bias[n];
#pragma unroll
  for (int v = 0; v < 8; ++v)
    C[(long)(mt * 16 + v + 8 * half) * VOCAB + n] = acc[v] + bn;
}

// beta[b*20+j] = tanh(vn[row] . beta_w + beta_b) * exp(0.01*(20-j))
__global__ void beta_kernel(const float* __restrict__ vn, const float* __restrict__ bw,
                            const float* __restrict__ bbp, float* __restrict__ beta)
{
  const int row = blockIdx.x;             // 0..1279
  const int t = threadIdx.x;              // 256
  float p = 0.0f;
  for (int v = t; v < VOCAB; v += 256) p += vn[(long)row * VOCAB + v] * bw[v];
  __shared__ float s[256];
  s[t] = p; __syncthreads();
  for (int w = 128; w > 0; w >>= 1) { if (t < w) s[t] += s[t + w]; __syncthreads(); }
  if (t == 0) {
    const int j = row % MV;
    beta[row] = tanhf(s[0] + bbp[0]) * __expf(0.01f * (float)(MV - j));
  }
}

// attn[b,v] = sum_j softmax_j(logits[b,j,v]) * beta[b,j]
__global__ void attn_kernel(const float* __restrict__ logits, const float* __restrict__ beta,
                            float* __restrict__ attn)
{
  const long idx = (long)blockIdx.x * 256 + threadIdx.x;
  if (idx >= (long)NBAT * VOCAB) return;
  const int b = (int)(idx >> 12);
  const int v = (int)(idx & (VOCAB - 1));
  float vals[MV], mx = -1e30f;
#pragma unroll
  for (int j = 0; j < MV; ++j) {
    vals[j] = logits[(long)(b * MV + j) * VOCAB + v];
    mx = fmaxf(mx, vals[j]);
  }
  float se = 0.0f, ws = 0.0f;
#pragma unroll
  for (int j = 0; j < MV; ++j) {
    const float e = __expf(vals[j] - mx);
    se += e; ws += e * beta[b * MV + j];
  }
  attn[idx] = ws / se;
}

// wr_tab[l*500+r] = e_table[r] . wr_w[l] + wr_b[l]
__global__ void wr_kernel(const float* __restrict__ etab, const float* __restrict__ wr_w,
                          const float* __restrict__ wr_b, float* __restrict__ wrt)
{
  const int i = blockIdx.x * 256 + threadIdx.x;
  if (i >= 1000) return;
  const int l = i / 500, r = i % 500;
  float s = 0.0f;
  for (int h = 0; h < HID; ++h) s += etab[r * HID + h] * wr_w[l * HID + h];
  wrt[i] = s + wr_b[l];
}

// X[i] = x_table[node_ids[i]]   (float4 rows)
__global__ void gather_x(const int* __restrict__ nids, const float* __restrict__ xtab,
                         float* __restrict__ X, int N)
{
  const long t = (long)blockIdx.x * 256 + threadIdx.x;
  const long node = t >> 5;
  if (node >= N) return;
  const int h = (int)(t & 31) * 4;
  *reinterpret_cast<float4*>(X + node * HID + h) =
      *reinterpret_cast<const float4*>(xtab + (long)nids[node] * HID + h);
}

// per-edge: msg = relu(x[src]*attn_e + w_rel*e_tab), atomic segment-sum into AGG[dst]
__global__ void edge_kernel(const int* __restrict__ src, const int* __restrict__ dst,
                            const int* __restrict__ eids, const int* __restrict__ nids,
                            const int* __restrict__ batch,
                            const float* __restrict__ attn, const float* __restrict__ wrt,
                            const float* __restrict__ etab,
                            const float* __restrict__ X, float* __restrict__ AGG, long E)
{
  const long t = (long)blockIdx.x * 256 + threadIdx.x;
  const long e = t >> 5;
  if (e >= E) return;
  const int h = (int)(t & 31) * 4;
  const int s = src[e], d = dst[e], r = eids[e];
  const float a = attn[(long)batch[s] * VOCAB + nids[s]];
  const float w = wrt[r];
  const float* xr = X + (long)s * HID + h;
  const float* er = etab + (long)r * HID + h;
  float* ar = AGG + (long)d * HID + h;
#pragma unroll
  for (int i = 0; i < 4; ++i) {
    const float m = fmaxf(fmaf(xr[i], a, w * er[i]), 0.0f);
    atomicAdd(&ar[i], m);                 // global_atomic_add_f32, L2-resident
  }
}

// global mean-pool numerators + counts
__global__ void pool_kernel(const float* __restrict__ X, const int* __restrict__ batch,
                            float* __restrict__ pool, float* __restrict__ cnt, int N)
{
  const long t = (long)blockIdx.x * 256 + threadIdx.x;
  const long node = t >> 5;
  if (node >= N) return;
  const int h = (int)(t & 31) * 4;
  const int b = batch[node];
  const float* xr = X + node * HID + h;
#pragma unroll
  for (int i = 0; i < 4; ++i) atomicAdd(&pool[(long)b * HID + h + i], xr[i]);
  if ((t & 31) == 0) atomicAdd(&cnt[b], 1.0f);
}

// xraw[b,n] = (ehr[b] @ node_emb)[n] / sum(ehr[b])
__global__ void xnode_kernel(const float* __restrict__ ehr, const float* __restrict__ nemb,
                             float* __restrict__ xraw)
{
  const int b = blockIdx.x;    // 64
  const int n = threadIdx.x;   // 128
  float s = 0.0f, tot = 0.0f;
  for (int k = 0; k < VOCAB; ++k) {
    const float ev = ehr[(long)b * VOCAB + k];
    tot += ev;
    s = fmaf(ev, nemb[(long)k * HID + n], s);
  }
  xraw[b * HID + n] = s / tot;
}

// concat [pool/count | x_node] -> cc[64,256]
__global__ void concat_kernel(const float* __restrict__ pool, const float* __restrict__ cnt,
                              const float* __restrict__ xnd, float* __restrict__ cc)
{
  const int i = blockIdx.x * 256 + threadIdx.x;
  if (i >= NBAT * 256) return;
  const int b = i >> 8, k = i & 255;
  cc[i] = (k < HID) ? pool[b * HID + k] / fmaxf(cnt[b], 1.0f)
                    : xnd[b * HID + (k - HID)];
}

extern "C" void kernel_launch(void* const* d_in, const int* in_sizes, int n_in,
                              void* d_out, int out_size, void* d_ws, size_t ws_size,
                              hipStream_t stream)
{
  const int*   node_ids   = (const int*)  d_in[0];
  const int*   edge_ids   = (const int*)  d_in[1];
  const int*   edge_index = (const int*)  d_in[2];
  const float* visit_node = (const float*)d_in[6];
  const float* ehr        = (const float*)d_in[7];
  const int*   batch      = (const int*)  d_in[8];
  const float* node_emb   = (const float*)d_in[10];
  const float* edge_emb   = (const float*)d_in[11];
  const float* lin_w      = (const float*)d_in[12];
  const float* lin_b      = (const float*)d_in[13];
  const float* alpha_w    = (const float*)d_in[14];
  const float* alpha_b    = (const float*)d_in[15];
  const float* beta_w     = (const float*)d_in[16];
  const float* beta_b     = (const float*)d_in[17];
  const float* conv_w     = (const float*)d_in[18];
  const float* conv_b     = (const float*)d_in[19];
  const float* wr_w       = (const float*)d_in[20];
  const float* wr_b       = (const float*)d_in[21];
  const float* eps        = (const float*)d_in[22];
  const float* mlp_w      = (const float*)d_in[23];
  const float* mlp_b      = (const float*)d_in[24];
  float* out = (float*)d_out;

  const int  N = in_sizes[0];     // 50000
  const long E = in_sizes[1];     // 800000
  const int* src = edge_index;
  const int* dst = edge_index + E;

  // workspace layout (floats); SCR doubles as alpha-logits then AGG (disjoint live ranges)
  float* ws   = (float*)d_ws;
  float* X0   = ws;                              // N*HID
  float* X1   = X0 + (long)N * HID;              // N*HID
  float* SCR  = X1 + (long)N * HID;              // max(N*HID, 1280*VOCAB)
  float* XTAB = SCR + (long)N * HID;             // 4096*128
  float* ETAB = XTAB + (long)VOCAB * HID;        // 500*128
  float* WRT  = ETAB + 500L * HID;               // 2*500
  float* ATTN = WRT + 1000;                      // 64*4096
  float* BETA = ATTN + (long)NBAT * VOCAB;       // 1280
  float* POOL = BETA + NBAT * MV;                // 64*128
  float* CNT  = POOL + NBAT * HID;               // 64
  float* XRAW = CNT + NBAT;                      // 64*128
  float* XND  = XRAW + NBAT * HID;               // 64*128
  float* CC   = XND + NBAT * HID;                // 64*256

  typedef void (*gemm_fn)(const float*, int, const float*, const float*,
                          const float*, int, const float*, float*, int, int, int, int);
  auto gemm = [&](gemm_fn kern, const float* A, int lda, const float* A2, const float* ep,
                  const float* B, int ldb, const float* bias,
                  float* C, int ldc, int M, int Nn, int K) {
    long tiles = (long)((M + 15) / 16) * ((Nn + 15) / 16);
    int blocks = (int)((tiles + 7) / 8);
    kern<<<blocks, 256, 0, stream>>>(A, lda, A2, ep, B, ldb, bias, C, ldc, M, Nn, K);
  };
  gemm_fn gemm_plain = gemm_f32_nt<false, false>;
  gemm_fn gemm_conv  = gemm_f32_nt<true,  true>;

  // --- tables: shared Linear applied once per vocab entry (f32 WMMA) ---
  gemm(gemm_plain, node_emb, HID, nullptr, nullptr, lin_w, HID, lin_b, XTAB, HID, VOCAB, HID, HID);
  gemm(gemm_plain, edge_emb, HID, nullptr, nullptr, lin_w, HID, lin_b, ETAB, HID, 500,   HID, HID);
  wr_kernel<<<4, 256, 0, stream>>>(ETAB, wr_w, wr_b, WRT);
  gather_x<<<(int)(((long)N * 32 + 255) / 256), 256, 0, stream>>>(node_ids, XTAB, X0, N);

  // --- two GNN layers ---
  for (int l = 0; l < 2; ++l) {
    const float* Xc = (l == 0) ? X0 : X1;
    float*       Xn = (l == 0) ? X1 : X0;
    // alpha logits: the 43-GFLOP GEMM -> bf16 WMMA + async-LDS A staging
    alpha_gemm_bf16<<<dim3(32, 80), 256, 0, stream>>>(
        visit_node, alpha_w + (long)l * VOCAB * VOCAB, alpha_b + (long)l * VOCAB, SCR);
    beta_kernel<<<NBAT * MV, 256, 0, stream>>>(visit_node, beta_w + (long)l * VOCAB,
                                               beta_b + l, BETA);
    attn_kernel<<<(NBAT * VOCAB) / 256, 256, 0, stream>>>(SCR, BETA, ATTN);
    // SCR now becomes AGG
    hipMemsetAsync(SCR, 0, (size_t)N * HID * sizeof(float), stream);
    edge_kernel<<<(int)((E * 32 + 255) / 256), 256, 0, stream>>>(
        src, dst, edge_ids, node_ids, batch, ATTN, WRT + l * 500, ETAB, Xc, SCR, E);
    // x = relu((agg + (1+eps)x) @ conv_w^T + conv_b)  (fused in A-loader)
    gemm(gemm_conv, SCR, HID, Xc, eps + l, conv_w + (long)l * HID * HID, HID,
         conv_b + (long)l * HID, Xn, HID, N, HID, HID);
  }
  float* XF = X0;  // after 2 layers result is back in X0

  // --- readout ---
  hipMemsetAsync(POOL, 0, (size_t)(NBAT * HID + NBAT) * sizeof(float), stream);
  pool_kernel<<<(int)(((long)N * 32 + 255) / 256), 256, 0, stream>>>(XF, batch, POOL, CNT, N);
  xnode_kernel<<<NBAT, HID, 0, stream>>>(ehr, node_emb, XRAW);
  gemm(gemm_plain, XRAW, HID, nullptr, nullptr, lin_w, HID, lin_b, XND, HID, NBAT, HID, HID);
  concat_kernel<<<NBAT, 256, 0, stream>>>(POOL, CNT, XND, CC);
  gemm(gemm_plain, CC, 2 * HID, nullptr, nullptr, mlp_w, 2 * HID, mlp_b, out, 25, NBAT, 25, 2 * HID);
}
